// DynamicResampler_v2_12232066859266
// MI455X (gfx1250) — compile-verified, matrix operand order checked by reading, not run on hardware
//
#include <hip/hip_runtime.h>
#include <hip/hip_bf16.h>

// ---------------------------------------------------------------------------
// DynamicResampler for MI455X (gfx1250, wave32).
// All matmuls via v_wmma_f32_16x16x32_bf16; K-tiles staged into LDS with
// global_load_async_to_lds_b128 (ASYNCcnt double-buffering), fragments read
// with ds_load_b128 in the CDNA5 16-bit A/B WMMA layout.
// Wave tile 64x(16*JT); block tile 128x(64*JT); JT=4 for the big GEMMs
// (1.0 ds_load_b128 per wmma), JT=2 for the N=128 attn@V GEMM.
// ---------------------------------------------------------------------------

typedef __bf16 bf16;
typedef __attribute__((ext_vector_type(16))) __bf16 v16bf;
typedef __attribute__((ext_vector_type(4)))  __bf16 bf16x4;
typedef __attribute__((ext_vector_type(8)))  float  v8f;
typedef __attribute__((ext_vector_type(4)))  unsigned int u32x4;

union FragB16 { v16bf v; u32x4 u[2]; };

#define E_DIM 4096L
#define H_DIM 32L
#define D_DIM 128L
#define M_DIM 256L
#define B_DIM 8L
#define N_DIM 256L

// ---------------------------------------------------------------------------
// Core WMMA GEMM:  C(128 x 64*JT tile) = A(rows x K) * B(cols x K)^T,
// bf16 in, f32 accum.  Block = 256 threads = 8 waves (2 in M x 4 in N);
// wave tile = 64(M) x 16*JT(N).  K staged 64 elements/stage, double buffered.
// ---------------------------------------------------------------------------
template <int JT, typename StoreF>
__device__ __forceinline__ void gemm_core(const bf16* __restrict__ A, long lda,
                                          const bf16* __restrict__ Bm, long ldb,
                                          int K, StoreF&& store)
{
    constexpr unsigned BN     = 64u * JT;        // 128 or 256 cols
    constexpr unsigned ABYTES = 128u * 128u;     // 128 rows * 64 bf16
    constexpr unsigned BBYTES = BN * 128u;       // BN rows * 64 bf16
    constexpr unsigned STAGE  = ABYTES + BBYTES;
    __shared__ __align__(16) unsigned char smem[2 * STAGE];

    const int t    = threadIdx.x;
    const int lane = t & 31;
    const int wave = t >> 5;
    const int wm   = wave >> 2;   // 0..1  (M direction, 64 rows each)
    const int wn   = wave & 3;    // 0..3  (N direction, 16*JT cols each)

    // LDS aperture: low 32 bits of a flat shared pointer are the LDS offset.
    const unsigned ldsBase = (unsigned)(unsigned long long)(size_t)&smem[0];

    // Stage one 128x64 A tile and BNx64 B tile: (4 + 2*JT) async b128/thread.
    auto issue = [&](int stage, int kpos) {
        const unsigned sb = ldsBase + (unsigned)stage * STAGE;
        const int r  = t >> 3;            // 0..31
        const int c8 = (t & 7) * 8;       // element offset within 64-wide K slab
#pragma unroll
        for (int j = 0; j < 4; ++j) {
            const int row = r + j * 32;
            unsigned long long ga =
                (unsigned long long)(size_t)(A + (size_t)row * lda + kpos + c8);
            asm volatile("global_load_async_to_lds_b128 %0, %1, off"
                         :: "v"(sb + (unsigned)(row * 128 + c8 * 2)), "v"(ga)
                         : "memory");
        }
#pragma unroll
        for (int j = 0; j < 2 * JT; ++j) {
            const int row = r + j * 32;
            unsigned long long gb =
                (unsigned long long)(size_t)(Bm + (size_t)row * ldb + kpos + c8);
            asm volatile("global_load_async_to_lds_b128 %0, %1, off"
                         :: "v"(sb + ABYTES + (unsigned)(row * 128 + c8 * 2)), "v"(gb)
                         : "memory");
        }
    };

    v8f acc[4][JT] = {};

    auto compute = [&](int s) {
        const unsigned char* sa  = smem + (s & 1) * STAGE;
        const unsigned char* sbp = sa + ABYTES;
#pragma unroll
        for (int c = 0; c < 2; ++c) {   // two 16x16x32 K-chunks per stage
            FragB16 af[4], bfr[JT];
            // CDNA5 16-bit A/B layout: lane<16 holds K 0..7 / 16..23,
            // lane>=16 holds K 8..15 / 24..31 (each half one b128 load).
            const int koffB = ((lane >> 4) * 8 + c * 32) * 2;
#pragma unroll
            for (int i = 0; i < 4; ++i) {
                const int row = wm * 64 + i * 16 + (lane & 15);
                const unsigned char* p = sa + row * 128 + koffB;
                af[i].u[0] = *(const u32x4*)(p);
                af[i].u[1] = *(const u32x4*)(p + 32);
            }
#pragma unroll
            for (int j = 0; j < JT; ++j) {
                const int col = wn * (16 * JT) + j * 16 + (lane & 15);
                const unsigned char* p = sbp + col * 128 + koffB;
                bfr[j].u[0] = *(const u32x4*)(p);
                bfr[j].u[1] = *(const u32x4*)(p + 32);
            }
#pragma unroll
            for (int i = 0; i < 4; ++i)
#pragma unroll
                for (int j = 0; j < JT; ++j)
                    acc[i][j] = __builtin_amdgcn_wmma_f32_16x16x32_bf16(
                        false, af[i].v, false, bfr[j].v,
                        (short)0, acc[i][j], false, false);
        }
    };

    const int nk = K >> 6;       // all K here are multiples of 128
    issue(0, 0);
    for (int s = 0; s < nk - 1; ++s) {
        issue((s + 1) & 1, (s + 1) << 6);
        if constexpr (JT == 2)
            asm volatile("s_wait_asynccnt 0x8" ::: "memory");  // stage s landed
        else
            asm volatile("s_wait_asynccnt 0xC" ::: "memory");
        __syncthreads();          // all waves' stage-s loads visible
        compute(s);
        __syncthreads();          // protect LDS buffer before overwrite
    }
    asm volatile("s_wait_asynccnt 0x0" ::: "memory");
    __syncthreads();
    compute(nk - 1);

    // C layout: VGPR v -> M = 16*i + v + 8*(lane>=16), N = 16*j + (lane&15)
#pragma unroll
    for (int i = 0; i < 4; ++i)
#pragma unroll
        for (int j = 0; j < JT; ++j)
#pragma unroll
            for (int v = 0; v < 8; ++v) {
                const int m = wm * 64 + i * 16 + v + ((lane >> 4) << 3);
                const int n = wn * (16 * JT) + j * 16 + (lane & 15);
                store(m, n, acc[i][j][v]);
            }
}

// ---------------------------------------------------------------------------
// GEMM wrappers
// ---------------------------------------------------------------------------
__global__ __launch_bounds__(256) void k_gemm_bias_bf16(
    const bf16* __restrict__ A, long lda, const bf16* __restrict__ W, long ldw,
    const float* __restrict__ bias, float scale,
    bf16* __restrict__ C, long ldc, int K)
{
    const long row0 = (long)blockIdx.x * 128, col0 = (long)blockIdx.y * 256;
    gemm_core<4>(A + row0 * lda, lda, W + col0 * ldw, ldw, K,
        [&](int m, int n, float v) {
            C[(row0 + m) * ldc + (col0 + n)] = (bf16)((v + bias[col0 + n]) * scale);
        });
}

// V projection, output scattered transposed per head: vT[b][h][d][n]
__global__ __launch_bounds__(256) void k_gemm_vproj(
    const bf16* __restrict__ A, const bf16* __restrict__ W,
    const float* __restrict__ bias, bf16* __restrict__ vT)
{
    const long row0 = (long)blockIdx.x * 128, col0 = (long)blockIdx.y * 256;
    gemm_core<4>(A + row0 * E_DIM, E_DIM, W + col0 * E_DIM, E_DIM, (int)E_DIM,
        [&](int m, int n, float v) {
            const long R = row0 + m, C_ = col0 + n;
            const long b = R >> 8, nn = R & 255;
            const long h = C_ >> 7, d = C_ & 127;
            vT[(((b * H_DIM + h) << 7) + d) * N_DIM + nn] = (bf16)(v + bias[C_]);
        });
}

// scores[b,h] = (q_h * k_{b,h}^T), scale already folded into q.  N tile = 256.
__global__ __launch_bounds__(256) void k_gemm_scores(
    const bf16* __restrict__ q, const bf16* __restrict__ k, float* __restrict__ s)
{
    const long z = blockIdx.z, h = z & 31, b = z >> 5;
    const long row0 = (long)blockIdx.x * 128;
    const bf16* At = q + row0 * E_DIM + h * D_DIM;
    const bf16* Bt = k + b * N_DIM * E_DIM + h * D_DIM;
    float* Ct = s + z * M_DIM * N_DIM;
    gemm_core<4>(At, E_DIM, Bt, E_DIM, (int)D_DIM,
        [&](int m, int n, float v) { Ct[(row0 + m) * N_DIM + n] = v; });
}

// o[b, m, h*128+d] = attn[b,h] * vT[b,h]^T   (N = 128 -> JT = 2)
__global__ __launch_bounds__(256) void k_gemm_av(
    const bf16* __restrict__ attn, const bf16* __restrict__ vT, bf16* __restrict__ o)
{
    const long z = blockIdx.z, h = z & 31, b = z >> 5;
    const long row0 = (long)blockIdx.x * 128;
    const bf16* At = attn + z * M_DIM * N_DIM + row0 * N_DIM;
    const bf16* Bt = vT + z * D_DIM * N_DIM;
    gemm_core<2>(At, N_DIM, Bt, N_DIM, (int)N_DIM,
        [&](int m, int n, float v) {
            o[(b * M_DIM + row0 + m) * E_DIM + h * D_DIM + n] = (bf16)v;
        });
}

__global__ __launch_bounds__(256) void k_gemm_out(
    const bf16* __restrict__ A, const bf16* __restrict__ W,
    const float* __restrict__ bias, float* __restrict__ out)
{
    const long row0 = (long)blockIdx.x * 128, col0 = (long)blockIdx.y * 256;
    gemm_core<4>(A + row0 * E_DIM, E_DIM, W + col0 * E_DIM, E_DIM, (int)E_DIM,
        [&](int m, int n, float v) {
            out[(row0 + m) * E_DIM + col0 + n] = v + bias[col0 + n];
        });
}

// ---------------------------------------------------------------------------
// LayerNorm (+optional pos-embed) producing bf16.  One block per row of 4096.
// out0 (optional): plain LN;  out1: LN + pos_embed[row & 255].
// ---------------------------------------------------------------------------
__global__ __launch_bounds__(256) void k_ln(
    const float* __restrict__ x, const float* __restrict__ w,
    const float* __restrict__ b, const float* __restrict__ pos,
    bf16* __restrict__ out0, bf16* __restrict__ out1)
{
    __shared__ float sm[16];
    const long row = blockIdx.x;
    const long pr  = row & 255;
    const float* xr = x + row * E_DIM;
    const int base = threadIdx.x * 16;

    float v[16];
    float s = 0.f, s2 = 0.f;
#pragma unroll
    for (int i = 0; i < 16; i += 4) {
        float4 f = *(const float4*)(xr + base + i);
        v[i] = f.x; v[i + 1] = f.y; v[i + 2] = f.z; v[i + 3] = f.w;
        s  += f.x + f.y + f.z + f.w;
        s2 += f.x * f.x + f.y * f.y + f.z * f.z + f.w * f.w;
    }
#pragma unroll
    for (int o = 16; o > 0; o >>= 1) {
        s  += __shfl_xor(s,  o, 32);
        s2 += __shfl_xor(s2, o, 32);
    }
    if ((threadIdx.x & 31) == 0) {
        sm[threadIdx.x >> 5]       = s;
        sm[8 + (threadIdx.x >> 5)] = s2;
    }
    __syncthreads();
    s = 0.f; s2 = 0.f;
#pragma unroll
    for (int i = 0; i < 8; ++i) { s += sm[i]; s2 += sm[8 + i]; }

    const float mean = s * (1.f / (float)E_DIM);
    const float var  = s2 * (1.f / (float)E_DIM) - mean * mean;
    const float r    = rsqrtf(var + 1e-5f);
#pragma unroll
    for (int i = 0; i < 16; ++i) {
        const int c = base + i;
        const float y = (v[i] - mean) * r * w[c] + b[c];
        if (out0) out0[row * E_DIM + c] = (bf16)y;
        out1[row * E_DIM + c] = (bf16)(y + pos[pr * E_DIM + c]);
    }
}

// f32 -> bf16 bulk convert (weights), float4 granularity, grid-stride
__global__ __launch_bounds__(256) void k_cvt(
    const float* __restrict__ src, bf16* __restrict__ dst, long n4)
{
    long i = (long)blockIdx.x * blockDim.x + threadIdx.x;
    const long stride = (long)gridDim.x * blockDim.x;
    for (; i < n4; i += stride) {
        float4 f = ((const float4*)src)[i];
        bf16x4 o;
        o[0] = (bf16)f.x; o[1] = (bf16)f.y; o[2] = (bf16)f.z; o[3] = (bf16)f.w;
        ((bf16x4*)dst)[i] = o;
    }
}

// Row softmax over 256 f32 scores -> bf16 attn.  One block per row.
__global__ __launch_bounds__(256) void k_softmax(
    const float* __restrict__ s, bf16* __restrict__ attn)
{
    __shared__ float sm[8];
    const long row = blockIdx.x;
    const float v = s[row * N_DIM + threadIdx.x];

    float m = v;
#pragma unroll
    for (int o = 16; o > 0; o >>= 1) m = fmaxf(m, __shfl_xor(m, o, 32));
    if ((threadIdx.x & 31) == 0) sm[threadIdx.x >> 5] = m;
    __syncthreads();
    float mx = sm[0];
#pragma unroll
    for (int i = 1; i < 8; ++i) mx = fmaxf(mx, sm[i]);
    __syncthreads();

    const float e = __expf(v - mx);
    float t = e;
#pragma unroll
    for (int o = 16; o > 0; o >>= 1) t += __shfl_xor(t, o, 32);
    if ((threadIdx.x & 31) == 0) sm[threadIdx.x >> 5] = t;
    __syncthreads();
    float sum = 0.f;
#pragma unroll
    for (int i = 0; i < 8; ++i) sum += sm[i];

    attn[row * N_DIM + threadIdx.x] = (bf16)(e / sum);
}

// attn mean over heads: out2[b,m,n] = (1/H) sum_h attn[b,h,m,n]
__global__ __launch_bounds__(256) void k_attn_mean(
    const bf16* __restrict__ attn, float* __restrict__ out2)
{
    const long idx = (long)blockIdx.x * 256 + threadIdx.x;   // < B*M*N
    const long b  = idx / (M_DIM * N_DIM);
    const long mn = idx % (M_DIM * N_DIM);
    float s = 0.f;
#pragma unroll
    for (int h = 0; h < 32; ++h)
        s += (float)attn[(b * H_DIM + h) * (M_DIM * N_DIM) + mn];
    out2[idx] = s * (1.f / 32.f);
}

// ---------------------------------------------------------------------------
extern "C" void kernel_launch(void* const* d_in, const int* in_sizes, int n_in,
                              void* d_out, int out_size, void* d_ws, size_t ws_size,
                              hipStream_t stream)
{
    const float* x    = (const float*)d_in[0];
    const float* qry  = (const float*)d_in[1];
    const float* pos  = (const float*)d_in[2];
    const float* lnqw = (const float*)d_in[3];
    const float* lnqb = (const float*)d_in[4];
    const float* lnkw = (const float*)d_in[5];
    const float* lnkb = (const float*)d_in[6];
    const float* Win  = (const float*)d_in[7];
    const float* bin  = (const float*)d_in[8];
    const float* Wout = (const float*)d_in[9];
    const float* bout = (const float*)d_in[10];
    float* out = (float*)d_out;

    char* ws = (char*)d_ws;
    size_t off = 0;
    auto take = [&](size_t bytes) -> void* {
        void* p = ws + off;
        off = (off + bytes + 255) & ~(size_t)255;
        return p;
    };

    const long BN = B_DIM * N_DIM;                 // 2048
    bf16*  kv      = (bf16*)take(BN * E_DIM * 2);          // LN(x)
    bf16*  kin     = (bf16*)take(BN * E_DIM * 2);          // LN(x)+pos
    bf16*  qin     = (bf16*)take(M_DIM * E_DIM * 2);       // LN(q)+pos
    bf16*  win_bf  = (bf16*)take(3 * E_DIM * E_DIM * 2);   // bf16 in_proj_w
    bf16*  wout_bf = (bf16*)take(E_DIM * E_DIM * 2);       // bf16 out_proj_w
    bf16*  qbuf    = (bf16*)take(M_DIM * E_DIM * 2);       // q / sqrt(D)
    bf16*  kbuf    = (bf16*)take(BN * E_DIM * 2);          // k
    bf16*  vT      = (bf16*)take(B_DIM * H_DIM * D_DIM * N_DIM * 2); // v transposed
    float* scores  = (float*)take(B_DIM * H_DIM * M_DIM * N_DIM * 4);
    bf16*  attn    = (bf16*)take(B_DIM * H_DIM * M_DIM * N_DIM * 2);
    bf16*  obuf    = (bf16*)take(BN * E_DIM * 2);          // attn@v, (B,M,E)
    (void)ws_size; (void)in_sizes; (void)n_in; (void)out_size;

    // 1) LayerNorms + pos-embed, bf16 outputs
    k_ln<<<(int)BN,    256, 0, stream>>>(x,   lnkw, lnkb, pos, kv, kin);
    k_ln<<<(int)M_DIM, 256, 0, stream>>>(qry, lnqw, lnqb, pos, (bf16*)nullptr, qin);

    // 2) Weights to bf16
    k_cvt<<<2048, 256, 0, stream>>>(Win,  win_bf,  3 * E_DIM * E_DIM / 4);
    k_cvt<<<2048, 256, 0, stream>>>(Wout, wout_bf, E_DIM * E_DIM / 4);

    // 3) Q/K/V projections (WMMA)
    const float inv_sqrt_d = 0.08838834764831845f;   // 1/sqrt(128)
    k_gemm_bias_bf16<<<dim3(2, 16, 1),  256, 0, stream>>>(
        qin, E_DIM, win_bf, E_DIM, bin, inv_sqrt_d, qbuf, E_DIM, (int)E_DIM);
    k_gemm_bias_bf16<<<dim3(16, 16, 1), 256, 0, stream>>>(
        kin, E_DIM, win_bf + E_DIM * E_DIM, E_DIM, bin + E_DIM, 1.f,
        kbuf, E_DIM, (int)E_DIM);
    k_gemm_vproj<<<dim3(16, 16, 1), 256, 0, stream>>>(
        kv, win_bf + 2 * E_DIM * E_DIM, bin + 2 * E_DIM, vT);

    // 4) scores = q.k^T (scale folded), softmax, o = attn.v
    k_gemm_scores<<<dim3(2, 1, 256), 256, 0, stream>>>(qbuf, kbuf, scores);
    k_softmax<<<(int)(B_DIM * H_DIM * M_DIM), 256, 0, stream>>>(scores, attn);
    k_gemm_av<<<dim3(2, 1, 256), 256, 0, stream>>>(attn, vT, obuf);

    // 5) out projection -> d_out[0 : B*M*E)
    k_gemm_out<<<dim3(16, 8, 1), 256, 0, stream>>>(obuf, wout_bf, bout, out);

    // 6) attn.mean over heads -> d_out[B*M*E : )
    k_attn_mean<<<(int)(B_DIM * M_DIM * N_DIM / 256), 256, 0, stream>>>(
        attn, out + B_DIM * M_DIM * E_DIM);
}
